// Ranker_6700148982312
// MI455X (gfx1250) — matrix operand order repaired and checked by reading, not compiled, not run on hardware
//
#include <hip/hip_runtime.h>
#include <hip/hip_bf16.h>
#include <math.h>
#include <stdint.h>

typedef __bf16 bf16;
typedef __attribute__((ext_vector_type(16))) __bf16 v16bf;
typedef __attribute__((ext_vector_type(8)))  __bf16 v8bf;
typedef __attribute__((ext_vector_type(8)))  float  v8f;
typedef __attribute__((ext_vector_type(4)))  int    v4i;

#define HDIM  4096
#define BATCH 64
#define SEQ   128
#define NGATE 16384   // 4H
#define K2H   8192    // 2H
#define MTOK  8192    // B*S
#define NHID  1024
#define KDATA 16384   // 4H concat features
#define EPSV  1e-7f

#ifndef __has_builtin
#define __has_builtin(x) 0
#endif

#if __has_builtin(__builtin_amdgcn_global_load_async_to_lds_b128)
#define ASYNC_STAGE 1
#else
#define ASYNC_STAGE 0
#endif

#define GLB_AS __attribute__((address_space(1)))
#define LDS_AS __attribute__((address_space(3)))

// ---------------------------------------------------------------- helpers
__device__ __forceinline__ float sigmf(float x) { return 1.0f / (1.0f + __expf(-x)); }

// async 16B global->LDS copy (per-lane), tracked by ASYNCcnt
__device__ __forceinline__ void async_copy16(const bf16* g, bf16* l) {
#if ASYNC_STAGE
    __builtin_amdgcn_global_load_async_to_lds_b128(
        (GLB_AS v4i*)(uintptr_t)g,
        (LDS_AS v4i*)(uint32_t)(uintptr_t)l, 0, 0);
#endif
}

__device__ __forceinline__ void wait_async_le1() {
#if ASYNC_STAGE
#if __has_builtin(__builtin_amdgcn_s_wait_asynccnt)
    __builtin_amdgcn_s_wait_asynccnt(1);
#else
    asm volatile("s_wait_asynccnt 0x1" ::: "memory");
#endif
#endif
}
__device__ __forceinline__ void wait_async_le0() {
#if ASYNC_STAGE
#if __has_builtin(__builtin_amdgcn_s_wait_asynccnt)
    __builtin_amdgcn_s_wait_asynccnt(0);
#else
    asm volatile("s_wait_asynccnt 0x0" ::: "memory");
#endif
#endif
}

// A fragment: 16x32 bf16 tile at (row0, k0) of row-major matrix A (lda elems).
// Contiguous k order (same as B): lanes 0-15 -> M rows, elements 0..15 = K kb..kb+15,
// kb = 0 for lanes 0-15, 16 for lanes 16-31.
__device__ __forceinline__ v16bf load_a_frag(const bf16* A, int lda, int row0, int k0) {
    int lane = threadIdx.x & 31;
    int m    = lane & 15;
    int kb   = (lane >> 4) ? 16 : 0;
    const bf16* p = A + (size_t)(row0 + m) * lda + (k0 + kb);
    v8bf lo = *(const v8bf*)(p);
    v8bf hi = *(const v8bf*)(p + 8);
    v16bf r;
#pragma unroll
    for (int i = 0; i < 8; ++i) { r[i] = lo[i]; r[i + 8] = hi[i]; }
    return r;
}

// B fragment from global: 32x16 bf16 tile = W^T slice, W row-major (N x K, ldw).
// B[k][n] = W[col0+n][k]; lanes 0-15 -> N cols K 0..15, lanes 16-31 -> K 16..31.
__device__ __forceinline__ v16bf load_b_frag(const bf16* W, int ldw, int k0, int col0) {
    int lane = threadIdx.x & 31;
    int n    = lane & 15;
    int ks   = (lane >> 4) ? 16 : 0;
    const bf16* p = W + (size_t)(col0 + n) * ldw + (k0 + ks);
    return *(const v16bf*)(p);
}

// B fragment from an LDS-staged tile laid out [64 cols][BSPAD k-elems]
#define BSPAD 40   // 80B row stride: conflict-free lane->bank mapping, 16B aligned
__device__ __forceinline__ v16bf load_b_frag_lds(const bf16* Bs, int colgrp) {
    int lane = threadIdx.x & 31;
    int n    = lane & 15;
    int ks   = (lane >> 4) ? 16 : 0;
    const bf16* p = Bs + (size_t)(colgrp * 16 + n) * BSPAD + ks;
    v8bf lo = *(const v8bf*)(p);
    v8bf hi = *(const v8bf*)(p + 8);
    v16bf r;
#pragma unroll
    for (int i = 0; i < 8; ++i) { r[i] = lo[i]; r[i + 8] = hi[i]; }
    return r;
}

__device__ __forceinline__ v8f wmma_bf16(v16bf a, v16bf b, v8f c) {
    return __builtin_amdgcn_wmma_f32_16x16x32_bf16(false, a, false, b, (short)0, c,
                                                   false, false);
}

// ---------------------------------------------------------------- prep kernels
__global__ void k_f32_to_bf16(const float* __restrict__ src, bf16* __restrict__ dst,
                              size_t n4) {
    size_t stride = (size_t)gridDim.x * blockDim.x;
    for (size_t i = (size_t)blockIdx.x * blockDim.x + threadIdx.x; i < n4; i += stride) {
        float4 v = ((const float4*)src)[i];
        size_t o = i * 4;
        dst[o + 0] = (bf16)v.x; dst[o + 1] = (bf16)v.y;
        dst[o + 2] = (bf16)v.z; dst[o + 3] = (bf16)v.w;
    }
}

__global__ void k_build_inp(const int* __restrict__ questions,
                            const int* __restrict__ contexts,
                            const int* __restrict__ masks,
                            const float* __restrict__ emb,
                            bf16* __restrict__ inp) {
    const size_t total4 = (size_t)MTOK * K2H / 4;
    size_t stride = (size_t)gridDim.x * blockDim.x;
    for (size_t i = (size_t)blockIdx.x * blockDim.x + threadIdx.x; i < total4; i += stride) {
        size_t idx = i * 4;
        int k2 = (int)(idx & (K2H - 1));
        int bs = (int)(idx >> 13);
        int b  = bs >> 7;
        bool valid = (masks[bs] == 0);
        int tok = (k2 < HDIM) ? contexts[bs] : questions[b];
        int col = k2 & (HDIM - 1);
        float4 v = *(const float4*)(emb + (size_t)tok * HDIM + col);
        if (!valid) { v.x = 0.f; v.y = 0.f; v.z = 0.f; v.w = 0.f; }
        inp[idx + 0] = (bf16)v.x; inp[idx + 1] = (bf16)v.y;
        inp[idx + 2] = (bf16)v.z; inp[idx + 3] = (bf16)v.w;
    }
}

__global__ void k_build_q(const int* __restrict__ questions,
                          const float* __restrict__ emb, float* __restrict__ Q) {
    int i = blockIdx.x * blockDim.x + threadIdx.x;
    if (i >= BATCH * HDIM) return;
    int b = i >> 12, j = i & (HDIM - 1);
    Q[i] = emb[(size_t)questions[b] * HDIM + j];
}

__global__ void k_lengths(const int* __restrict__ masks, float* __restrict__ len) {
    int b = threadIdx.x;
    if (b >= BATCH) return;
    int c = 0;
    for (int s = 0; s < SEQ; ++s) c += (masks[b * SEQ + s] == 0) ? 1 : 0;
    len[b] = (float)c;
}

__global__ void k_zero_state(float* __restrict__ h, float* __restrict__ c,
                             float* __restrict__ ps, bf16* __restrict__ hbf) {
    int i = blockIdx.x * blockDim.x + threadIdx.x;
    if (i >= BATCH * HDIM) return;
    h[i] = 0.f; c[i] = 0.f; ps[i] = 0.f; hbf[i] = (bf16)0.f;
}

// ---------------------------------------------------------------- big GEMM
// xW = inp(8192x8192,bf16) @ W_ih^T + b_ih -> fp32 (8192x16384)
// Block: 256 thr (8 waves) -> 256x64 tile; wave -> 32x64 (2x4 accum tiles).
// B tile (64 cols x 32 K, 4KB) staged to LDS via async copies, double-buffered:
// kills the 8x intra-block redundancy on the 256MB (non-L2-resident) W_ih.
__global__ void k_gemm_xw(const bf16* __restrict__ inp, const bf16* __restrict__ Wih,
                          const float* __restrict__ b_ih, float* __restrict__ xW) {
    __shared__ bf16 Bs[2][64][BSPAD];
    int wave = threadIdx.x >> 5;
    int lane = threadIdx.x & 31;
    int row0 = blockIdx.y * 256 + wave * 32;
    int col0 = blockIdx.x * 64;

    // staging map: thread i copies 16B; col = i>>2, k-offset = (i&3)*8
    int tcol = threadIdx.x >> 2;
    int tsub = (threadIdx.x & 3) * 8;
    const bf16* gB = Wih + (size_t)(col0 + tcol) * K2H + tsub;

    v8f acc[2][4] = {};
    const int KSTEPS = K2H / 32;

#if ASYNC_STAGE
    async_copy16(gB, &Bs[0][tcol][tsub]);
#endif

#pragma unroll 1
    for (int kb = 0; kb < KSTEPS; ++kb) {
        int k = kb * 32;
#if ASYNC_STAGE
        int buf = kb & 1;
        if (kb + 1 < KSTEPS) {
            async_copy16(gB + k + 32, &Bs[buf ^ 1][tcol][tsub]);
            wait_async_le1();          // current buffer's copies complete
        } else {
            wait_async_le0();
        }
        __syncthreads();               // visible to all waves
#else
        int buf = 0;
        v8bf tmp = *(const v8bf*)(gB + k);
        __syncthreads();               // previous compute done before overwrite
        *(v8bf*)&Bs[0][tcol][tsub] = tmp;
        __syncthreads();
#endif
        v16bf a0 = load_a_frag(inp, K2H, row0,      k);
        v16bf a1 = load_a_frag(inp, K2H, row0 + 16, k);
#pragma unroll
        for (int j = 0; j < 4; ++j) {
            v16bf b = load_b_frag_lds(&Bs[buf][0][0], j);
            acc[0][j] = wmma_bf16(a0, b, acc[0][j]);
            acc[1][j] = wmma_bf16(a1, b, acc[1][j]);
        }
        __syncthreads();               // all waves done reading before next stage lands
    }

    int mrow = (lane >> 4) * 8;
#pragma unroll
    for (int s = 0; s < 2; ++s) {
#pragma unroll
        for (int j = 0; j < 4; ++j) {
#pragma unroll
            for (int v = 0; v < 8; ++v) {
                int m = row0 + s * 16 + mrow + v;
                int n = col0 + j * 16 + (lane & 15);
                float out = acc[s][j][v] + b_ih[n];
                // streaming store: keep L2 free for the L2-resident W_hh scan
                __builtin_nontemporal_store(out, &xW[(size_t)m * NGATE + n]);
            }
        }
    }
}

// ---------------------------------------------------------------- LSTM step
// One launch per timestep t. Block owns 16 columns of H; computes all 4 gates
// for all 64 batch rows (16 WMMA tiles, 2 per wave), then fused update in LDS.
// W_hh (bf16, 128MB) stays resident in the 192MB L2 across all 128 steps.
__global__ void k_lstm_step(const float* __restrict__ xW, const bf16* __restrict__ Whh,
                            const float* __restrict__ b_hh, const int* __restrict__ masks,
                            int t, float* __restrict__ h, float* __restrict__ c,
                            bf16* __restrict__ hbf, float* __restrict__ ps) {
    __shared__ float gbuf[4][BATCH][16];
    int wave = threadIdx.x >> 5;
    int lane = threadIdx.x & 31;
    int j0 = blockIdx.x * 16;

#pragma unroll
    for (int tt = 0; tt < 2; ++tt) {
        int tid = wave * 2 + tt;       // 0..15
        int g   = tid >> 2;            // gate 0..3
        int mt  = tid & 3;             // m-tile 0..3
        int row0 = mt * 16;
        int col  = g * HDIM + j0;
        v8f acc = {};
#pragma unroll 1
        for (int k = 0; k < HDIM; k += 32) {
            v16bf a = load_a_frag(hbf, HDIM, row0, k);
            v16bf b = load_b_frag(Whh, HDIM, k, col);
            acc = wmma_bf16(a, b, acc);
        }
        int mrow = (lane >> 4) * 8;
#pragma unroll
        for (int v = 0; v < 8; ++v)
            gbuf[g][row0 + mrow + v][lane & 15] = acc[v];
    }
    __syncthreads();

    for (int e = threadIdx.x; e < BATCH * 16; e += blockDim.x) {
        int b = e >> 4, n = e & 15;
        int j = j0 + n;
        size_t xrow = (size_t)(b * SEQ + t) * NGATE;
        float gi = gbuf[0][b][n] + xW[xrow + 0 * HDIM + j] + b_hh[0 * HDIM + j];
        float gf = gbuf[1][b][n] + xW[xrow + 1 * HDIM + j] + b_hh[1 * HDIM + j];
        float gg = gbuf[2][b][n] + xW[xrow + 2 * HDIM + j] + b_hh[2 * HDIM + j];
        float go = gbuf[3][b][n] + xW[xrow + 3 * HDIM + j] + b_hh[3 * HDIM + j];
        if (masks[b * SEQ + t] == 0) {
            float cn = sigmf(gf) * c[(size_t)b * HDIM + j] + sigmf(gi) * tanhf(gg);
            float hn = sigmf(go) * tanhf(cn);
            c[(size_t)b * HDIM + j] = cn;
            h[(size_t)b * HDIM + j] = hn;
            ps[(size_t)b * HDIM + j] += hn;
            hbf[(size_t)b * HDIM + j] = (bf16)hn;
        }
    }
}

// ---------------------------------------------------------------- MLP
__global__ void k_build_data(const float* __restrict__ Q, const float* __restrict__ ps,
                             const float* __restrict__ len, bf16* __restrict__ data) {
    int i = blockIdx.x * blockDim.x + threadIdx.x;
    if (i >= BATCH * KDATA) return;
    int b = i >> 14, k = i & (KDATA - 1);
    int chunk = k >> 12, j = k & (HDIM - 1);
    float q = Q[(size_t)b * HDIM + j];
    float p = ps[(size_t)b * HDIM + j] / len[b];
    float v = (chunk == 0) ? q : (chunk == 1) ? p : (chunk == 2) ? (q - p) : (q * p);
    data[i] = (bf16)v;
}

__global__ void k_gemm_hid(const bf16* __restrict__ data, const bf16* __restrict__ W1b,
                           const float* __restrict__ b1, float* __restrict__ hid) {
    int wave = threadIdx.x >> 5;
    int tile = blockIdx.x * 8 + wave;      // 0..255
    int mt = tile & 3, nt = tile >> 2;
    int row0 = mt * 16, col0 = nt * 16;
    v8f acc = {};
#pragma unroll 1
    for (int k = 0; k < KDATA; k += 32) {
        v16bf a = load_a_frag(data, KDATA, row0, k);
        v16bf b = load_b_frag(W1b, KDATA, k, col0);
        acc = wmma_bf16(a, b, acc);
    }
    int lane = threadIdx.x & 31;
    int mrow = (lane >> 4) * 8;
#pragma unroll
    for (int v = 0; v < 8; ++v) {
        int m = row0 + mrow + v, n = col0 + (lane & 15);
        float x = acc[v] + b1[n];
        hid[(size_t)m * NHID + n] = (x > 0.f) ? x : 0.01f * x;
    }
}

__global__ void k_loss(const float* __restrict__ hid, const float* __restrict__ W2,
                       const float* __restrict__ b2, const int* __restrict__ labels,
                       float* __restrict__ out) {
    __shared__ float ls[8];
    int wave = threadIdx.x >> 5;
    int lane = threadIdx.x & 31;
    float wsum = 0.f;
    for (int b = wave; b < BATCH; b += 8) {
        float dot = 0.f;
        for (int k = lane; k < NHID; k += 32)
            dot += hid[(size_t)b * NHID + k] * W2[k];
#pragma unroll
        for (int off = 16; off > 0; off >>= 1)
            dot += __shfl_down(dot, off, 32);
        if (lane == 0) {
            float y = (float)labels[b];
            float s = sigmf(dot + b2[0]);
            float p = fminf(fmaxf(s, EPSV), 1.0f - EPSV);
            float bce = -(y * logf(p) + (1.0f - y) * logf(1.0f - p));
            float pt = expf(-bce);
            wsum += (1.0f - pt) * (1.0f - pt) * bce;
        }
    }
    if (lane == 0) ls[wave] = wsum;
    __syncthreads();
    if (threadIdx.x == 0) {
        float tot = 0.f;
        for (int w = 0; w < 8; ++w) tot += ls[w];
        out[0] = tot;
    }
}

// ---------------------------------------------------------------- launch
extern "C" void kernel_launch(void* const* d_in, const int* in_sizes, int n_in,
                              void* d_out, int out_size, void* d_ws, size_t ws_size,
                              hipStream_t stream) {
    const int*   questions = (const int*)d_in[0];
    const int*   contexts  = (const int*)d_in[1];
    const int*   masks     = (const int*)d_in[2];
    const int*   labels    = (const int*)d_in[3];
    const float* emb       = (const float*)d_in[4];
    const float* W_ih      = (const float*)d_in[5];
    const float* W_hh      = (const float*)d_in[6];
    const float* b_ih      = (const float*)d_in[7];
    const float* b_hh      = (const float*)d_in[8];
    const float* W1        = (const float*)d_in[9];
    const float* b1        = (const float*)d_in[10];
    const float* W2        = (const float*)d_in[11];
    const float* b2        = (const float*)d_in[12];

    char* ws = (char*)d_ws;
    size_t off = 0;
    float* xW   = (float*)(ws + off); off += (size_t)MTOK * NGATE * 4;     // 512 MiB
    bf16*  Wihb = (bf16*)(ws + off);  off += (size_t)NGATE * K2H * 2;      // 256 MiB
    bf16*  Whhb = (bf16*)(ws + off);  off += (size_t)NGATE * HDIM * 2;     // 128 MiB (L2-resident)
    bf16*  inpb = (bf16*)(ws + off);  off += (size_t)MTOK * K2H * 2;       // 128 MiB
    bf16*  W1b  = (bf16*)(ws + off);  off += (size_t)NHID * KDATA * 2;     // 32 MiB
    bf16*  datb = (bf16*)(ws + off);  off += (size_t)BATCH * KDATA * 2;
    float* Q    = (float*)(ws + off); off += (size_t)BATCH * HDIM * 4;
    float* h    = (float*)(ws + off); off += (size_t)BATCH * HDIM * 4;
    float* c    = (float*)(ws + off); off += (size_t)BATCH * HDIM * 4;
    float* ps   = (float*)(ws + off); off += (size_t)BATCH * HDIM * 4;
    bf16*  hbf  = (bf16*)(ws + off);  off += (size_t)BATCH * HDIM * 2;
    float* hid  = (float*)(ws + off); off += (size_t)BATCH * NHID * 4;
    float* len  = (float*)(ws + off); off += (size_t)BATCH * 4;

    k_f32_to_bf16<<<4096, 256, 0, stream>>>(W_ih, Wihb, (size_t)NGATE * K2H / 4);
    k_f32_to_bf16<<<4096, 256, 0, stream>>>(W_hh, Whhb, (size_t)NGATE * HDIM / 4);
    k_f32_to_bf16<<<2048, 256, 0, stream>>>(W1,   W1b,  (size_t)NHID * KDATA / 4);

    k_build_inp<<<32768, 256, 0, stream>>>(questions, contexts, masks, emb, inpb);
    k_build_q<<<(BATCH * HDIM + 255) / 256, 256, 0, stream>>>(questions, emb, Q);
    k_lengths<<<1, 64, 0, stream>>>(masks, len);
    k_zero_state<<<(BATCH * HDIM + 255) / 256, 256, 0, stream>>>(h, c, ps, hbf);

    // input projection GEMM (2.2 TFLOP), LDS-staged, async double-buffered
    {
        dim3 grid(NGATE / 64, MTOK / 256);
        k_gemm_xw<<<grid, 256, 0, stream>>>(inpb, Wihb, b_ih, xW);
    }

    // sequential LSTM scan: 128 launches, W_hh (bf16, 128 MiB) stays in L2
    for (int t = 0; t < SEQ; ++t)
        k_lstm_step<<<HDIM / 16, 256, 0, stream>>>(xW, Whhb, b_hh, masks, t,
                                                   h, c, hbf, ps);

    k_build_data<<<(BATCH * KDATA + 255) / 256, 256, 0, stream>>>(Q, ps, len, datb);
    k_gemm_hid<<<32, 256, 0, stream>>>(datb, W1b, b1, hid);
    k_loss<<<1, 256, 0, stream>>>(hid, W2, b2, labels, (float*)d_out);
}
